// SimpleTEMEncoder_32195074851234
// MI455X (gfx1250) — matrix operand order repaired and checked by reading, not compiled, not run on hardware
//
#include <hip/hip_runtime.h>
#include <cstdint>

// ---------------------------------------------------------------------------
// Problem constants (from reference)
// ---------------------------------------------------------------------------
#define NB     16384
#define NG     32
#define ND     256
#define NA     256
#define NH     512
#define BD     ((size_t)NB * ND)

typedef __attribute__((ext_vector_type(16))) _Float16 f16x16;
typedef __attribute__((ext_vector_type(8)))  _Float16 f16x8;
typedef __attribute__((ext_vector_type(8)))  float    f32x8;

// Workspace layout (bytes)
//   WBloc : GEMM1 B-frags : KC=8,  NT=16 -> 4096 recs * 32B = 128KB
//   WB1   : GEMM2 B-frags : KC=8,  NT=32 -> 8192 recs * 32B = 256KB
//   WB2   : GEMM3 B-frags : KC=16, NT=16 -> 8192 recs * 32B = 256KB
//   biasLoc[256] f32, spvsl[256] f32
#define OFF_WBLOC   0
#define OFF_WB1     (128 * 1024)
#define OFF_WB2     (OFF_WB1 + 256 * 1024)
#define OFF_BIASLOC (OFF_WB2 + 256 * 1024)
#define OFF_SPVSL   (OFF_BIASLOC + 1024)

__device__ __forceinline__ float softplus_f(float x) {
    return (x > 20.f) ? x : log1pf(__expf(x));
}

// ---------------------------------------------------------------------------
// Prep: pack weights into f16 WMMA B-fragment order.
// B-frag record r = ((kc*NT + nt)*32 + lane):
//   lane n = nt*16 + (lane&15); kbase = kc*32 + ((lane&16)?8:0)
//   element e of v16h = W[n][kbase + (e<8 ? e : e+8)]   (two 8-runs, A table order)
// ---------------------------------------------------------------------------
__device__ __forceinline__ void pack_bfrag(f16x16* out, int rec, const float* W,
                                           int rowlen, int NT) {
    int lane = rec & 31;
    int t    = rec >> 5;
    int nt   = t % NT;
    int kc   = t / NT;
    int n    = nt * 16 + (lane & 15);
    int kb   = kc * 32 + ((lane & 16) ? 8 : 0);
    const float* src = W + (size_t)n * rowlen + kb;
    union { f16x16 v; _Float16 e[16]; } u;
#pragma unroll
    for (int e = 0; e < 16; ++e) u.e[e] = (_Float16)src[(e < 8) ? e : e + 8];
    out[rec] = u.v;
}

__global__ __launch_bounds__(256)
void prep_kernel(const float* __restrict__ Wloc, const float* __restrict__ W1,
                 const float* __restrict__ W2, const float* __restrict__ vsl,
                 char* __restrict__ ws) {
    int tid = blockIdx.x * blockDim.x + threadIdx.x;
    f16x16* wbloc = (f16x16*)(ws + OFF_WBLOC);
    f16x16* wb1   = (f16x16*)(ws + OFF_WB1);
    f16x16* wb2   = (f16x16*)(ws + OFF_WB2);
    float* biasLoc = (float*)(ws + OFF_BIASLOC);
    float* spvsl   = (float*)(ws + OFF_SPVSL);
    if (tid < 4096) {
        pack_bfrag(wbloc, tid, Wloc, NA + 1, 16);          // delta = action @ Wloc^T
    } else if (tid < 12288) {
        pack_bfrag(wb1, tid - 4096, W1, ND, 32);           // h = delta @ W1^T
    } else if (tid < 20480) {
        pack_bfrag(wb2, tid - 12288, W2, NH, 16);          // s = h @ W2^T
    } else if (tid < 20736) {
        int j = tid - 20480;
        biasLoc[j] = Wloc[(size_t)j * (NA + 1) + NA];      // "+1" scale column
        spvsl[j]   = softplus_f(vsl[j]);
    }
}

// ---------------------------------------------------------------------------
// Fused GEMM chain: delta -> mean, h=relu(.), sd=0.1+softplus(vsl)*softplus(.)
// 64 threads = 2 waves, 16 batch rows per wave, 32 rows per block.
// All matmuls via v_wmma_f32_16x16x32_f16 (f32 accumulate).
// ---------------------------------------------------------------------------
#define RS_D 264   // delta LDS row stride (halfs): 256 + 8 pad
#define RS_H 520   // h     LDS row stride (halfs): 512 + 8 pad
#define WAVE_LDS (16 * RS_D + 16 * RS_H)   // halfs per wave = 12544

__device__ __forceinline__ f16x16 lds_afrag(const _Float16* row, int kc, int hi) {
    // A-matrix 16x32 f16 layout: lane holds two contiguous 8-half runs:
    //   [kc*32 + hi*8, +8) and [kc*32 + 16 + hi*8, +8)
    const _Float16* p = row + kc * 32 + hi * 8;
    union { f16x16 v; f16x8 h[2]; } u;
    u.h[0] = *(const f16x8*)(p);
    u.h[1] = *(const f16x8*)(p + 16);
    return u.v;
}

__global__ __launch_bounds__(64)
void gemm_chain_kernel(const float* __restrict__ lastloc,
                       const float* __restrict__ action,
                       const float* __restrict__ b1,
                       const float* __restrict__ b2,
                       const char* __restrict__ ws,
                       float* __restrict__ out) {
    __shared__ __align__(16) _Float16 smem[2 * WAVE_LDS];

    const f16x16* wbloc = (const f16x16*)(ws + OFF_WBLOC);
    const f16x16* wb1   = (const f16x16*)(ws + OFF_WB1);
    const f16x16* wb2   = (const f16x16*)(ws + OFF_WB2);
    const float* biasLoc = (const float*)(ws + OFF_BIASLOC);
    const float* spvsl   = (const float*)(ws + OFF_SPVSL);

    const int lane = threadIdx.x & 31;
    const int w    = threadIdx.x >> 5;
    const int hi   = lane >> 4;        // half-wave select
    const int mlo  = lane & 15;
    const int rowBase = blockIdx.x * 32 + w * 16;

    _Float16* dA = smem + w * WAVE_LDS;       // 16 x RS_D (f16 delta tile)
    _Float16* hA = dA + 16 * RS_D;            // 16 x RS_H (f16 h tile)

    // ---- GEMM1 A-fragments straight from global action (f32 -> f16) ----
    f16x16 aA[8];
#pragma unroll
    for (int kc = 0; kc < 8; ++kc) {
        const float* ap = action + (size_t)(rowBase + mlo) * NA + kc * 32 + hi * 8;
        union { float4 q[4]; float f[16]; } s;
        s.q[0] = *(const float4*)(ap);
        s.q[1] = *(const float4*)(ap + 4);
        s.q[2] = *(const float4*)(ap + 16);
        s.q[3] = *(const float4*)(ap + 20);
        union { f16x16 v; _Float16 e[16]; } u;
#pragma unroll
        for (int e = 0; e < 16; ++e) u.e[e] = (_Float16)s.f[e];
        aA[kc] = u.v;
    }

    // ---- GEMM1: delta = action @ Wloc[:, :256]^T  (+ bias column) ----
    for (int nt = 0; nt < 16; ++nt) {
        f32x8 acc = {};
#pragma unroll
        for (int kc = 0; kc < 8; ++kc) {
            f16x16 b = wbloc[((size_t)(kc * 16 + nt) * 32) + lane];
            acc = __builtin_amdgcn_wmma_f32_16x16x32_f16(
                false, aA[kc], false, b, (short)0, acc, false, false);
        }
        const int n  = nt * 16 + mlo;
        const float bb = biasLoc[n];
#pragma unroll
        for (int j = 0; j < 8; ++j) {
            const int m = j + hi * 8;
            const float dv = acc[j] + bb;                 // delta
            dA[m * RS_D + n] = (_Float16)dv;
            const size_t gi = (size_t)(rowBase + m) * ND + n;
            out[BD + gi] = lastloc[gi] + dv;              // mean
        }
    }
    __syncthreads();

    // ---- GEMM2: h = relu(delta @ W1^T + b1) ----
    f16x16 aD[8];
#pragma unroll
    for (int kc = 0; kc < 8; ++kc) aD[kc] = lds_afrag(dA + mlo * RS_D, kc, hi);

    for (int nt = 0; nt < 32; ++nt) {
        f32x8 acc = {};
#pragma unroll
        for (int kc = 0; kc < 8; ++kc) {
            f16x16 b = wb1[((size_t)(kc * 32 + nt) * 32) + lane];
            acc = __builtin_amdgcn_wmma_f32_16x16x32_f16(
                false, aD[kc], false, b, (short)0, acc, false, false);
        }
        const int n = nt * 16 + mlo;
        const float bias = b1[n];
#pragma unroll
        for (int j = 0; j < 8; ++j) {
            const int m = j + hi * 8;
            float hv = acc[j] + bias;
            hv = hv > 0.f ? hv : 0.f;
            hA[m * RS_H + n] = (_Float16)hv;
        }
    }
    __syncthreads();

    // ---- GEMM3: sd = 0.1 + softplus(vsl) * softplus(h @ W2^T + b2) ----
    f16x16 aH[16];
#pragma unroll
    for (int kc = 0; kc < 16; ++kc) aH[kc] = lds_afrag(hA + mlo * RS_H, kc, hi);

    for (int nt = 0; nt < 16; ++nt) {
        f32x8 acc = {};
#pragma unroll
        for (int kc = 0; kc < 16; ++kc) {
            f16x16 b = wb2[((size_t)(kc * 16 + nt) * 32) + lane];
            acc = __builtin_amdgcn_wmma_f32_16x16x32_f16(
                false, aH[kc], false, b, (short)0, acc, false, false);
        }
        const int n = nt * 16 + mlo;
        const float bias = b2[n];
        const float sv   = spvsl[n];
#pragma unroll
        for (int j = 0; j < 8; ++j) {
            const int m = j + hi * 8;
            const float sd = 0.1f + sv * softplus_f(acc[j] + bias);
            out[2 * BD + (size_t)(rowBase + m) * ND + n] = sd;
        }
    }
}

// ---------------------------------------------------------------------------
// Select-location: single HBM pass over (B,G,D) guesses/sds (1 GB -> ~45us
// floor at 23.3 TB/s). Guesses staged in LDS for the weighted-sum reuse.
// One block (256 threads, 8 waves) per batch row.
// ---------------------------------------------------------------------------
__device__ __forceinline__ uint32_t pcg_hash(uint32_t x) {
    x = x * 747796405u + 2891336453u;
    uint32_t w = ((x >> ((x >> 28u) + 4u)) ^ x) * 277803737u;
    return (w >> 22u) ^ w;
}

__device__ __forceinline__ float gauss_hash(uint32_t i) {
    uint32_t a = pcg_hash(2u * i + 1u);
    uint32_t c = pcg_hash(2u * i + 2u);
    float u1 = (float)(a >> 8) * (1.0f / 16777216.0f) + 1e-7f;
    float u2 = (float)(c >> 8) * (1.0f / 16777216.0f);
    return sqrtf(-2.0f * __logf(u1)) * __cosf(6.2831853071795864f * u2);
}

__global__ __launch_bounds__(256)
void select_kernel(const float* __restrict__ guesses,
                   const float* __restrict__ lsds,
                   const int* __restrict__ numfound,
                   float* __restrict__ out) {
    __shared__ float gS[NG * ND];     // 32 KB guesses stash
    __shared__ float meanS[ND], sdS[ND], red[ND];
    __shared__ float matchS[NG], wS[NG];

    const int b    = blockIdx.x;
    const int t    = threadIdx.x;
    const int lane = t & 31;
    const int w    = t >> 5;

    meanS[t] = out[BD + (size_t)b * ND + t];
    sdS[t]   = out[2 * BD + (size_t)b * ND + t];
    __syncthreads();

    const int nf = numfound[b];

    // Pass 1: per-guess Gaussian log-density (g-dependent term only; the
    // per-b constants cancel in softmax). Wave w handles g = w, w+8, w+16, w+24.
#pragma unroll
    for (int giter = 0; giter < 4; ++giter) {
        const int g = w + giter * 8;
        const float* gp = guesses + ((size_t)b * NG + g) * ND;
        const float* sp = lsds    + ((size_t)b * NG + g) * ND;
        float partial = 0.f;
#pragma unroll
        for (int i = 0; i < 8; ++i) {
            const int d = lane + i * 32;
            const float v = gp[d];
            const float s = sp[d];
            gS[g * ND + d] = v;
            const float den = s + sdS[d] + 1e-6f;
            const float td  = (v - meanS[d]) / den;
            partial += td * td;
        }
#pragma unroll
        for (int off = 16; off > 0; off >>= 1)
            partial += __shfl_xor(partial, off, 32);
        if (lane == 0) matchS[g] = -0.5f * partial;
    }
    __syncthreads();

    // Softmax over G=32 in wave 0 (mask g >= num_found).
    if (w == 0) {
        float m = (lane < nf) ? matchS[lane] : -3.4e38f;
        float mx = m;
#pragma unroll
        for (int off = 16; off > 0; off >>= 1)
            mx = fmaxf(mx, __shfl_xor(mx, off, 32));
        float e = (lane < nf) ? __expf(m - mx) : 0.f;
        float ssum = e;
#pragma unroll
        for (int off = 16; off > 0; off >>= 1)
            ssum += __shfl_xor(ssum, off, 32);
        wS[lane] = (ssum > 0.f) ? (e / ssum) : 0.f;
    }
    __syncthreads();

    // Pass 2: weighted sum from LDS stash (thread t owns column d = t).
    float integ = 0.f;
#pragma unroll
    for (int g = 0; g < NG; ++g) integ += wS[g] * gS[g * ND + t];

    const float diff = (integ - meanS[t]) / sdS[t];
    red[t] = diff * diff;
    __syncthreads();
    for (int s = 128; s > 0; s >>= 1) {
        if (t < s) red[t] += red[t + s];
        __syncthreads();
    }
    const bool toofar = red[0] > 4.0f;                    // ||.|| > 2
    const bool cond   = toofar || (integ != integ) || (nf == 0);

    const float noise    = gauss_hash((uint32_t)(b * ND + t));
    const float expected = meanS[t] + noise * sdS[t];
    out[(size_t)b * ND + t] = cond ? expected : integ;
}

// ---------------------------------------------------------------------------
extern "C" void kernel_launch(void* const* d_in, const int* in_sizes, int n_in,
                              void* d_out, int out_size, void* d_ws, size_t ws_size,
                              hipStream_t stream) {
    const float* lastloc = (const float*)d_in[0];
    const float* action  = (const float*)d_in[1];
    const float* guesses = (const float*)d_in[2];
    const float* lsds    = (const float*)d_in[3];
    const int*   nf      = (const int*)d_in[4];
    const float* Wloc    = (const float*)d_in[5];
    const float* W1      = (const float*)d_in[6];
    const float* b1      = (const float*)d_in[7];
    const float* W2      = (const float*)d_in[8];
    const float* b2      = (const float*)d_in[9];
    const float* vsl     = (const float*)d_in[10];
    float* out = (float*)d_out;
    char*  ws  = (char*)d_ws;

    prep_kernel<<<81, 256, 0, stream>>>(Wloc, W1, W2, vsl, ws);
    gemm_chain_kernel<<<NB / 32, 64, 0, stream>>>(lastloc, action, b1, b2, ws, out);
    select_kernel<<<NB, 256, 0, stream>>>(guesses, lsds, nf, out);
}
